// ATPModule_38062000177838
// MI455X (gfx1250) — matrix operand order repaired and checked by reading, not compiled, not run on hardware
//
#include <hip/hip_runtime.h>
#include <hip/hip_fp16.h>
#include <math.h>

typedef __attribute__((ext_vector_type(16))) _Float16 v16h;
typedef __attribute__((ext_vector_type(8)))  float    v8f;

#define B_   2
#define H_   32
#define S_   1024
#define D_   4096
#define LV   576
#define GRID_ 24
#define FDIM 256

// ---------------- kernel 0: zero the ws accumulator (every call) ----------------
__global__ void zero_ws_kernel(float* acc) {
  int i = blockIdx.x * blockDim.x + threadIdx.x;
  if (i < B_ * 2 * LV) acc[i] = 0.f;
}

// ---------------- kernel 1: big bandwidth-bound reduction ----------------
// acc layout: [b][region(0=self,1=cross)][k] , k in [0, LV)
__global__ __launch_bounds__(288) void reduce_attn_kernel(
    const float* __restrict__ attn, float* __restrict__ acc) {
  int blk   = blockIdx.x;
  int chunk = blk & 15;        // 16 chunks of 64 rows per (b,h): 9 self + 7 cross
  int bh    = blk >> 4;        // 0..63
  int b     = bh >> 5;
  int region, q0;
  if (chunk < 9) { region = 0; q0 = chunk * 64; }            // q in [0,576)
  else           { region = 1; q0 = LV + (chunk - 9) * 64; } // q in [576,1024)

  int k2 = threadIdx.x;        // 0..287 -> covers k = 2*k2, 2*k2+1 (coalesced float2)
  const float* base = attn + ((size_t)bh * S_ + (size_t)q0) * S_;

  float ax = 0.f, ay = 0.f;
  #pragma unroll 4
  for (int q = 0; q < 64; ++q) {
    const float2* row = (const float2*)(base + (size_t)q * S_);
    float2 v = row[k2];
    ax += v.x;
    ay += v.y;
  }
  float* dst = acc + ((b * 2 + region) * LV) + 2 * k2;
  atomicAdd(dst,     ax);
  atomicAdd(dst + 1, ay);
}

// ---------------- kernel 2: pool + WMMA MLP + masks ----------------
__device__ inline float sigmoidf_(float x) { return 1.f / (1.f + __expf(-x)); }

__global__ __launch_bounds__(512) void mlp_masks_kernel(
    const float* __restrict__ acc,
    const float* __restrict__ W1, const float* __restrict__ b1,
    const float* __restrict__ Wr, const float* __restrict__ br,
    const float* __restrict__ Ws, const float* __restrict__ bs,
    float* __restrict__ out_masks)   // out_masks -> mask_r[2*576], then mask_s, then mask
{
  __shared__ float    s_self[B_][LV];
  __shared__ float    s_cross[B_][LV];
  __shared__ _Float16 poolA[16][2 * FDIM];   // 16x512 f16, rows 2..15 zero
  __shared__ float    act[16][FDIM];         // relu(pool @ W1 + b1)
  __shared__ float    theta[4];              // theta_r[0..1], theta_s[0..1]

  const int tid = threadIdx.x;

  // ---- means from the accumulated sums ----
  for (int i = tid; i < B_ * LV; i += 512) {
    int b = i / LV, k = i % LV;
    s_self [b][k] = acc[(b * 2 + 0) * LV + k] * (1.f / (float)(H_ * LV));
    s_cross[b][k] = acc[(b * 2 + 1) * LV + k] * (1.f / (float)(H_ * (S_ - LV)));
  }
  // ---- zero the padded activation tile ----
  for (int i = tid; i < 16 * 2 * FDIM; i += 512)
    ((_Float16*)poolA)[i] = (_Float16)0.f;
  __syncthreads();

  // ---- adaptive avg pool (Lv=576 -> 256) for self and cross, rows 0..1 ----
  for (int i = tid; i < B_ * 2 * FDIM; i += 512) {
    int b = i / (2 * FDIM), j = i % (2 * FDIM);
    int p  = (j < FDIM) ? j : j - FDIM;
    int st = (p * LV) / FDIM;
    int en = ((p + 1) * LV + FDIM - 1) / FDIM;
    const float* src = (j < FDIM) ? s_self[b] : s_cross[b];
    float s = 0.f;
    for (int t = st; t < en; ++t) s += src[t];
    poolA[b][j] = (_Float16)(s / (float)(en - st));
  }
  __syncthreads();

  // ---- act = relu(poolA(16x512) @ W1(512x256) + b1) via v_wmma_f32_16x16x32_f16 ----
  // 16 waves, wave w owns output N-tile [w*16, w*16+16)
  const int wave   = tid >> 5;
  const int lane   = tid & 31;
  const int mrow   = lane & 15;
  const int khalfA = (lane < 16) ? 0 : 8;    // A: 16-bit 16x32 layout
  const int koffB  = (lane < 16) ? 0 : 16;   // B: 16-bit 32x16 layout
  const int ncol   = wave * 16 + (lane & 15);

  v8f c = {};
  for (int ks = 0; ks < 2 * FDIM; ks += 32) {
    v16h a, bf;
    #pragma unroll
    for (int j = 0; j < 8; ++j) {
      a[j]     = poolA[mrow][ks + 2 * khalfA + j];
      a[j + 8] = poolA[mrow][ks + 2 * khalfA + 8 + j];
    }
    #pragma unroll
    for (int j = 0; j < 16; ++j)
      bf[j] = (_Float16)W1[(size_t)(ks + koffB + j) * FDIM + ncol];
    c = __builtin_amdgcn_wmma_f32_16x16x32_f16(
        /*neg_a=*/false, a, /*neg_b=*/false, bf,
        /*c_mod=*/(short)0, c, /*reuse_a=*/false, /*reuse_b=*/false);
  }
  #pragma unroll
  for (int r = 0; r < 8; ++r) {            // C/D layout: lanes0-15 M=r, lanes16-31 M=r+8
    int row = (lane < 16) ? r : r + 8;
    float v = c[r] + b1[ncol];
    act[row][ncol] = v > 0.f ? v : 0.f;
  }
  __syncthreads();

  // ---- two 256->1 heads + sigmoid thresholds ----
  if (tid < B_) {
    float sr = 0.f, ss = 0.f;
    for (int i = 0; i < FDIM; ++i) {
      float a_ = act[tid][i];
      sr += a_ * Wr[i];
      ss += a_ * Ws[i];
    }
    theta[tid]     = sigmoidf_(sr + br[0]);
    theta[2 + tid] = sigmoidf_(ss + bs[0]);
  }
  __syncthreads();

  // ---- soft masks ----
  for (int i = tid; i < B_ * LV; i += 512) {
    int b = i / LV, k = i % LV;
    float sred = 0.5f * (s_self[b][k] + s_cross[b][k]);
    float mr = sigmoidf_((sred - theta[b]) * 100.f);
    int row = k / GRID_, col = k % GRID_;
    float sspat = (((row & 1) == 0) && ((col & 1) == 0)) ? 0.25f : -100.f;
    float ms = sigmoidf_((sspat - theta[2 + b]) * 100.f);
    out_masks[i]              = mr;
    out_masks[B_ * LV + i]    = ms;
    out_masks[2 * B_ * LV + i]= fmaxf(mr, ms);
  }
}

extern "C" void kernel_launch(void* const* d_in, const int* in_sizes, int n_in,
                              void* d_out, int out_size, void* d_ws, size_t ws_size,
                              hipStream_t stream) {
  const float* hs   = (const float*)d_in[0];
  const float* attn = (const float*)d_in[1];
  const void*  pid  = d_in[2];
  // d_in[3] = num_vision_tokens (576, fixed by setup)
  const float* W1 = (const float*)d_in[4];
  const float* b1 = (const float*)d_in[5];
  const float* Wr = (const float*)d_in[6];
  const float* br = (const float*)d_in[7];
  const float* Ws = (const float*)d_in[8];
  const float* bs = (const float*)d_in[9];

  float* out = (float*)d_out;
  float* acc = (float*)d_ws;   // needs B_*2*LV*4 = 9216 bytes

  // pass-throughs: hidden_states then position_ids (raw int32 bits)
  const size_t hsN = (size_t)B_ * S_ * D_;          // 8,388,608 floats
  hipMemcpyAsync(out, hs, hsN * sizeof(float), hipMemcpyDeviceToDevice, stream);
  hipMemcpyAsync(out + hsN, pid, (size_t)in_sizes[2] * sizeof(int),
                 hipMemcpyDeviceToDevice, stream);

  // scores + MLP + masks
  zero_ws_kernel<<<(B_ * 2 * LV + 255) / 256, 256, 0, stream>>>(acc);
  reduce_attn_kernel<<<B_ * H_ * 16, 288, 0, stream>>>(attn, acc);

  float* mask_base = out + hsN + (size_t)in_sizes[2];
  mlp_masks_kernel<<<1, 512, 0, stream>>>(acc, W1, b1, Wr, br, Ws, bs, mask_base);
}
